// TUCKERGBM_84138409329265
// MI455X (gfx1250) — matrix operand order repaired and checked by reading, not compiled, not run on hardware
//
#include <hip/hip_runtime.h>
#include <hip/hip_bf16.h>

typedef __attribute__((ext_vector_type(4)))  _Float16 v4h;
typedef __attribute__((ext_vector_type(8)))  _Float16 v8h;
typedef __attribute__((ext_vector_type(16))) _Float16 v16h;
typedef __attribute__((ext_vector_type(8)))  float    v8f;

#define BATCH   2048
#define NUMIN   8192
#define NUMOUT  8192
#define NUMMAP  4096
#define RANK    64
#define KSPLIT  4

// ---------------------------------------------------------------------------
// Fragment helpers (layouts per CDNA5 ISA §7.12.2, wave32)
// A 16x32 f16 tile: lane L<16 -> row L, K {0..7}U{16..23}; lane>=16 -> row L-16,
// K {8..15}U{24..31}.  B 32x16: stored N-major (Bt[n*ldb+k]); lane%16 = col,
// lane/16 selects K {0..7}U{16..23}/{8..15}U{24..31} as for A; contiguous in k -> b128 loads.
// ---------------------------------------------------------------------------
__device__ __forceinline__ v16h frag16(const _Float16* lo, const _Float16* hi) {
    v8h a = *(const v8h*)lo;
    v8h b = *(const v8h*)hi;
    v16h r;
#pragma unroll
    for (int i = 0; i < 8; ++i) { r[i] = a[i]; r[i + 8] = b[i]; }
    return r;
}

__device__ __forceinline__ v8f wmma_f16(v16h a, v16h b, v8f c) {
    return __builtin_amdgcn_wmma_f32_16x16x32_f16(false, a, false, b, (short)0, c,
                                                  false, false);
}

// Fast sigmoid: v_exp_f32 + v_rcp_f32 (avoid IEEE div_scale/div_fixup chain).
__device__ __forceinline__ float sigmoidf_(float x) {
    return __builtin_amdgcn_rcpf(1.0f + __expf(-x));
}

// ---------------------------------------------------------------------------
// K0: one-time weight preparation.
//  - wxf/wyf/whf (K x 64 f32) -> f16 row-major copies (B-layout for expand GEMMs)
//  - and f16 transposed copies WT[n*K+k] (B-layout for projection GEMMs)
//  - core (64^3) -> coreT1[(j*64+k)*64+i]  (Bt for U_x = AX @ core_i(jk))
//               -> coreT2[(i*64+k)*64+j]  (Bt for U_y = AY2 @ core_j(ik))
// ---------------------------------------------------------------------------
__global__ void prep_weights(const float* __restrict__ wxf,
                             const float* __restrict__ wyf,
                             const float* __restrict__ whf,
                             const float* __restrict__ core,
                             _Float16* wxfT, _Float16* wyfT, _Float16* whfT,
                             _Float16* wxfh, _Float16* wyfh, _Float16* whfh,
                             _Float16* coreT1, _Float16* coreT2) {
    int idx = blockIdx.x * blockDim.x + threadIdx.x;
    if (idx < NUMIN * RANK) {
        int k = idx >> 6, n = idx & 63;
        _Float16 x = (_Float16)wxf[idx];
        wxfh[idx] = x;
        wxfT[(size_t)n * NUMIN + k] = x;
        _Float16 y = (_Float16)wyf[idx];
        wyfh[idx] = y;
        wyfT[(size_t)n * NUMOUT + k] = y;
    }
    if (idx < NUMMAP * RANK) {
        int k = idx >> 6, n = idx & 63;
        _Float16 h = (_Float16)whf[idx];
        whfh[idx] = h;
        whfT[(size_t)n * NUMMAP + k] = h;
    }
    if (idx < RANK * RANK * RANK) {
        int i = idx >> 12, j = (idx >> 6) & 63, kk = idx & 63;
        _Float16 c = (_Float16)core[idx];
        coreT1[(size_t)(j * 64 + kk) * 64 + i] = c;
        coreT2[(size_t)(i * 64 + kk) * 64 + j] = c;
    }
}

// ---------------------------------------------------------------------------
// K1: factor projection  P_part[kz] = A(B x K) @ W(K x 64)  (K-slice partial)
// A is f32 (raw inputs/outputs) or f16 (staged activations); WT is 64 x K f16
// (N-major).  grid = (128 M-tiles, KSPLIT), block = 128 (4 waves, one N-tile
// of 16 each).  A tile (16x64) staged via LDS with on-the-fly f32->f16 cvt;
// two independent WMMA accumulator chains per iteration (no RAW hazard).
// No atomics: each K-slice writes its own partial buffer (deterministic).
// ---------------------------------------------------------------------------
template <bool AHALF>
__global__ void proj_gemm(const void* __restrict__ A_,
                          const _Float16* __restrict__ WT,
                          float* __restrict__ Ppart, int K) {
    __shared__ __align__(16) _Float16 lds_a[16 * 64];
    const int m0   = blockIdx.x * 16;
    const int kz   = blockIdx.y;
    const int KS   = K / KSPLIT;
    const int kbeg = kz * KS, kend = kbeg + KS;
    const int wave = threadIdx.x >> 5, lane = threadIdx.x & 31;
    const int n0   = wave * 16;

    v8f acc0 = {}, acc1 = {};
    const int e  = threadIdx.x * 8;       // this thread's 8 staged elements
    const int r  = e >> 6;                // row in tile
    const int kk = e & 63;                // k in tile

    const int row = lane & 15;
    const int kh  = (lane >> 4) * 8;      // A-frag K half select
    const int ks  = (lane >> 4) * 8;      // B-frag K half select
    const _Float16* bbase = WT + (size_t)(n0 + (lane & 15)) * K;

    for (int k0 = kbeg; k0 < kend; k0 += 64) {
        if (AHALF) {
            const _Float16* src = (const _Float16*)A_ + (size_t)(m0 + r) * K + k0 + kk;
            *(v8h*)&lds_a[e] = *(const v8h*)src;
            if (k0 + 64 < kend) __builtin_prefetch((const char*)(src + 64), 0, 1);
        } else {
            const float* src = (const float*)A_ + (size_t)(m0 + r) * K + k0 + kk;
            float4 f0 = *(const float4*)src;
            float4 f1 = *(const float4*)(src + 4);
            v8h h;
            h[0] = (_Float16)f0.x; h[1] = (_Float16)f0.y;
            h[2] = (_Float16)f0.z; h[3] = (_Float16)f0.w;
            h[4] = (_Float16)f1.x; h[5] = (_Float16)f1.y;
            h[6] = (_Float16)f1.z; h[7] = (_Float16)f1.w;
            *(v8h*)&lds_a[e] = h;
            if (k0 + 64 < kend) __builtin_prefetch((const char*)(src + 64), 0, 1);
        }
        __syncthreads();

        v16h af0 = frag16(&lds_a[row * 64 + kh], &lds_a[row * 64 + kh + 16]);
        v16h af1 = frag16(&lds_a[row * 64 + 32 + kh], &lds_a[row * 64 + 32 + kh + 16]);
        const _Float16* bp = bbase + k0 + ks;
        v16h bf0 = frag16(bp, bp + 16);
        v16h bf1 = frag16(bp + 32, bp + 48);
        acc0 = wmma_f16(af0, bf0, acc0);
        acc1 = wmma_f16(af1, bf1, acc1);
        __syncthreads();
    }
    v8f acc = acc0 + acc1;

    float* P = Ppart + (size_t)kz * BATCH * RANK;
    const int col   = n0 + (lane & 15);
    const int rbase = m0 + ((lane >> 4) << 3);
#pragma unroll
    for (int p = 0; p < 8; ++p) P[(size_t)(rbase + p) * RANK + col] = acc[p];
}

// K1b: reduce KSPLIT partials -> f32 (for einsum contraction) and/or f16 (WMMA A).
template <bool WF, bool WH>
__global__ void reduce_partials(const float* __restrict__ Ppart,
                                float* __restrict__ Pf,
                                _Float16* __restrict__ Ph) {
    int idx = blockIdx.x * blockDim.x + threadIdx.x;
    if (idx >= BATCH * RANK) return;
    float s = 0.f;
#pragma unroll
    for (int z = 0; z < KSPLIT; ++z) s += Ppart[(size_t)z * BATCH * RANK + idx];
    if (WF) Pf[idx] = s;
    if (WH) Ph[idx] = (_Float16)s;
}

// ---------------------------------------------------------------------------
// K2: expansion GEMM  out = act(A(B x 64 f16) @ Bt(N x 64 f16)^T + bias)
// K=64 -> two independent WMMA K-steps, fragments loaded straight from global
// (contiguous b128).  Also used (BIAS=ACT=false) for U = AX @ core (N=4096).
// grid = (N/64, 128), block = 128 (4 waves, one 16x16 tile each).
// All epilogue options are template params -> branch-free epilogue.
// ---------------------------------------------------------------------------
template <bool BIAS, bool ACT, bool O32, bool O16>
__global__ void expand_gemm(const _Float16* __restrict__ Ah,
                            const _Float16* __restrict__ Bt,
                            const float* __restrict__ bias,
                            float* __restrict__ out32,
                            _Float16* __restrict__ out16, int N) {
    const int wave = threadIdx.x >> 5, lane = threadIdx.x & 31;
    const int m0 = blockIdx.y * 16;
    const int n0 = blockIdx.x * 64 + wave * 16;

    const int row = m0 + (lane & 15);
    const int kh  = (lane >> 4) * 8;
    const _Float16* ap = Ah + (size_t)row * RANK;
    v16h a0 = frag16(ap + kh, ap + kh + 16);
    v16h a1 = frag16(ap + 32 + kh, ap + 32 + kh + 16);

    const int n  = n0 + (lane & 15);
    const int ks = (lane >> 4) * 8;
    const _Float16* bp = Bt + (size_t)n * RANK;
    v16h b0 = frag16(bp + ks, bp + ks + 16);
    v16h b1 = frag16(bp + 32 + ks, bp + 32 + ks + 16);

    v8f z = {};
    v8f acc0 = wmma_f16(a0, b0, z);
    v8f acc1 = wmma_f16(a1, b1, z);
    v8f acc  = acc0 + acc1;

    const int   rbase = m0 + ((lane >> 4) << 3);
    const float bv    = BIAS ? bias[n] : 0.f;
#pragma unroll
    for (int p = 0; p < 8; ++p) {
        float v = acc[p] + bv;
        if (ACT) v = sigmoidf_(v);
        size_t o = (size_t)(rbase + p) * N + n;
        if (O32) out32[o] = v;
        if (O16) out16[o] = (_Float16)v;
    }
}

// ---------------------------------------------------------------------------
// K3: einsum second contraction against precomputed U (B x 4096 f32).
//  MODE 0: c[b,k] = sum_j S[b,j] * U[b, j*64+k]   ('bi,bj,ijk->bk' tail)
//  MODE 1: c[b,j] = sum_k S[b,k] * U[b, j*64+k]   ('bi,bk,ijk->bj' tail)
// One wave per batch row; each lane owns 2 output indices.
// ---------------------------------------------------------------------------
template <int MODE>
__global__ void einsum_contract(const float* __restrict__ S,
                                const float* __restrict__ U,
                                float* __restrict__ Cf,
                                _Float16* __restrict__ Ch) {
    const int wave = threadIdx.x >> 5, lane = threadIdx.x & 31;
    const int b = blockIdx.x * 4 + wave;
    const float* u = U + (size_t)b * (RANK * RANK);
    const float* s = S + (size_t)b * RANK;
    float a0 = 0.f, a1 = 0.f;
    const int o0 = lane, o1 = lane + 32;
#pragma unroll 4
    for (int t = 0; t < RANK; ++t) {
        float sv = s[t];
        if (MODE == 0) {
            a0 = fmaf(sv, u[t * 64 + o0], a0);
            a1 = fmaf(sv, u[t * 64 + o1], a1);
        } else {
            a0 = fmaf(sv, u[o0 * 64 + t], a0);
            a1 = fmaf(sv, u[o1 * 64 + t], a1);
        }
    }
    Cf[(size_t)b * RANK + o0] = a0;
    Cf[(size_t)b * RANK + o1] = a1;
    Ch[(size_t)b * RANK + o0] = (_Float16)a0;
    Ch[(size_t)b * RANK + o1] = (_Float16)a1;
}

// ---------------------------------------------------------------------------
extern "C" void kernel_launch(void* const* d_in, const int* in_sizes, int n_in,
                              void* d_out, int out_size, void* d_ws, size_t ws_size,
                              hipStream_t stream) {
    const float* inputs  = (const float*)d_in[0];
    const float* outputs = (const float*)d_in[1];
    const float* core    = (const float*)d_in[2];
    const float* wxf     = (const float*)d_in[3];
    const float* wyf     = (const float*)d_in[4];
    const float* whf     = (const float*)d_in[5];
    const float* ib      = (const float*)d_in[6];
    const float* ob      = (const float*)d_in[7];
    const float* hb      = (const float*)d_in[8];

    float* out_x = (float*)d_out;                               // x_probs (B x 8192)
    float* out_y = out_x + (size_t)BATCH * NUMIN;               // y_probs (B x 8192)

    // ---- carve workspace ----
    char*  ws  = (char*)d_ws;
    size_t off = 0;
    auto alloc = [&](size_t bytes) -> void* {
        void* p = ws + off;
        off += (bytes + 255) & ~(size_t)255;
        return p;
    };
    _Float16* wxfT  = (_Float16*)alloc((size_t)RANK * NUMIN * 2);
    _Float16* wyfT  = (_Float16*)alloc((size_t)RANK * NUMOUT * 2);
    _Float16* whfT  = (_Float16*)alloc((size_t)RANK * NUMMAP * 2);
    _Float16* wxfh  = (_Float16*)alloc((size_t)NUMIN * RANK * 2);
    _Float16* wyfh  = (_Float16*)alloc((size_t)NUMOUT * RANK * 2);
    _Float16* whfh  = (_Float16*)alloc((size_t)NUMMAP * RANK * 2);
    _Float16* coreT1= (_Float16*)alloc((size_t)RANK * RANK * RANK * 2);
    _Float16* coreT2= (_Float16*)alloc((size_t)RANK * RANK * RANK * 2);
    float*    Ppart = (float*)   alloc((size_t)KSPLIT * BATCH * RANK * 4);
    _Float16* AXh   = (_Float16*)alloc((size_t)BATCH * RANK * 2);
    float*    AYf   = (float*)   alloc((size_t)BATCH * RANK * 4);
    float*    AHf   = (float*)   alloc((size_t)BATCH * RANK * 4);
    float*    AY2f  = (float*)   alloc((size_t)BATCH * RANK * 4);
    _Float16* AY2h  = (_Float16*)alloc((size_t)BATCH * RANK * 2);
    float*    AH2f  = (float*)   alloc((size_t)BATCH * RANK * 4);
    float*    C1f   = (float*)   alloc((size_t)BATCH * RANK * 4);
    _Float16* C1h   = (_Float16*)alloc((size_t)BATCH * RANK * 2);
    float*    C2f   = (float*)   alloc((size_t)BATCH * RANK * 4);
    _Float16* C2h   = (_Float16*)alloc((size_t)BATCH * RANK * 2);
    float*    C3f   = (float*)   alloc((size_t)BATCH * RANK * 4);
    _Float16* C3h   = (_Float16*)alloc((size_t)BATCH * RANK * 2);
    float*    C4f   = (float*)   alloc((size_t)BATCH * RANK * 4);
    _Float16* C4h   = (_Float16*)alloc((size_t)BATCH * RANK * 2);
    float*    U     = (float*)   alloc((size_t)BATCH * RANK * RANK * 4); // 32MB, reused
    _Float16* Hh    = (_Float16*)alloc((size_t)BATCH * NUMMAP * 2);      // 16MB, reused
    _Float16* Yh    = (_Float16*)alloc((size_t)BATCH * NUMOUT * 2);      // 32MB
    (void)ws_size; (void)n_in; (void)in_sizes; (void)out_size;

    const dim3 blk(128);
    const dim3 gProj(BATCH / 16, KSPLIT);
    const dim3 gExpOut(NUMOUT / 64, BATCH / 16);
    const dim3 gExpMap(NUMMAP / 64, BATCH / 16);
    const int  gRed = (BATCH * RANK) / 256;

    // K0: weight prep
    prep_weights<<<(NUMIN * RANK) / 256, 256, 0, stream>>>(
        wxf, wyf, whf, core, wxfT, wyfT, whfT, wxfh, wyfh, whfh, coreT1, coreT2);

    // ax = inputs @ wxf  (f16 only; invariant for the whole chain)
    proj_gemm<false><<<gProj, blk, 0, stream>>>(inputs, wxfT, Ppart, NUMIN);
    reduce_partials<false, true><<<gRed, 256, 0, stream>>>(Ppart, nullptr, AXh);

    // ay = outputs @ wyf
    proj_gemm<false><<<gProj, blk, 0, stream>>>(outputs, wyfT, Ppart, NUMOUT);
    reduce_partials<true, false><<<gRed, 256, 0, stream>>>(Ppart, AYf, nullptr);

    // U_x[b, j*64+k] = sum_i ax[b,i] core[i,j,k]   (reused by 3 einsums)
    expand_gemm<false, false, true, false><<<gExpMap, blk, 0, stream>>>(
        AXh, coreT1, nullptr, U, nullptr, RANK * RANK);

    // h = sigmoid( (sum_j ay_j U_x[:,j,:]) @ whf^T + hb )
    einsum_contract<0><<<BATCH / 4, blk, 0, stream>>>(AYf, U, C1f, C1h);
    expand_gemm<true, true, false, true><<<gExpMap, blk, 0, stream>>>(
        C1h, whfh, hb, nullptr, Hh, NUMMAP);

    // ah = h @ whf ; y_probs = sigmoid( (sum_k ah_k U_x[:,:,k]) @ wyf^T + ob )
    proj_gemm<true><<<gProj, blk, 0, stream>>>(Hh, whfT, Ppart, NUMMAP);
    reduce_partials<true, false><<<gRed, 256, 0, stream>>>(Ppart, AHf, nullptr);
    einsum_contract<1><<<BATCH / 4, blk, 0, stream>>>(AHf, U, C2f, C2h);
    expand_gemm<true, true, true, true><<<gExpOut, blk, 0, stream>>>(
        C2h, wyfh, ob, out_y, Yh, NUMOUT);

    // h2 = hidprobs(inputs, y_probs): ay2 = y @ wyf, reuse U_x
    proj_gemm<true><<<gProj, blk, 0, stream>>>(Yh, wyfT, Ppart, NUMOUT);
    reduce_partials<true, true><<<gRed, 256, 0, stream>>>(Ppart, AY2f, AY2h);
    einsum_contract<0><<<BATCH / 4, blk, 0, stream>>>(AY2f, U, C3f, C3h);
    expand_gemm<true, true, false, true><<<gExpMap, blk, 0, stream>>>(
        C3h, whfh, hb, nullptr, Hh, NUMMAP);

    // ah2 = h2 @ whf
    proj_gemm<true><<<gProj, blk, 0, stream>>>(Hh, whfT, Ppart, NUMMAP);
    reduce_partials<true, false><<<gRed, 256, 0, stream>>>(Ppart, AH2f, nullptr);

    // U_y[b, i*64+k] = sum_j ay2[b,j] core[i,j,k]  (overwrites U; last U_x use done)
    expand_gemm<false, false, true, false><<<gExpMap, blk, 0, stream>>>(
        AY2h, coreT2, nullptr, U, nullptr, RANK * RANK);

    // x_probs = sigmoid( (sum_k ah2_k U_y[:,:,k]) @ wxf^T + ib )
    einsum_contract<1><<<BATCH / 4, blk, 0, stream>>>(AH2f, U, C4f, C4h);
    expand_gemm<true, true, true, false><<<gExpOut, blk, 0, stream>>>(
        C4h, wxfh, ib, out_x, nullptr, NUMIN);
}
